// GModel_24988119728845
// MI455X (gfx1250) — matrix operand order repaired and checked by reading, not compiled, not run on hardware
//
#include <hip/hip_runtime.h>

typedef _Float16 v8h  __attribute__((ext_vector_type(8)));
typedef _Float16 v16h __attribute__((ext_vector_type(16)));
typedef float    v8f  __attribute__((ext_vector_type(8)));

#define NCELL 2048
#define NDRUG 1024
#define DGENE 4096
#define DCNA  2048
#define DMUT  2048
#define DDRUG 2048
#define SDIM  128
#define ADIM  32
#define NTOT  3072
#define KTOP  10

union HFrag { v16h v; v8h h[2]; };

// ---------------------------------------------------------------------------
// WMMA GEMM: C[M,N] = X[M,K] * Y[N,K]^T   (f16 in, f32 out).
// M % 128 == 0, N % 128 == 0, K % 32 == 0.
// Block = 256 threads = 8 waves; block tile 128x128; wave tile 32x64
// (2 A-frags x 4 B-frags = 8 WMMAs per 32-K step, software-pipelined
// global->LDS staging so VMEM latency hides behind the WMMA chain).
// ---------------------------------------------------------------------------
__global__ __launch_bounds__(256)
void gemm_xyt_f16(const _Float16* __restrict__ X, const _Float16* __restrict__ Y,
                  float* __restrict__ C, int M, int N, int K,
                  int ldx, int ldy, int ldc) {
  __shared__ __align__(16) _Float16 As[128 * 32];
  __shared__ __align__(16) _Float16 Bs[128 * 32];
  const int bm = blockIdx.y * 128;
  const int bn = blockIdx.x * 128;
  const int tid  = threadIdx.x;
  const int wave = tid >> 5;            // 0..7
  const int lane = tid & 31;
  const int hi   = lane >> 4;           // 0: lanes 0-15, 1: lanes 16-31
  const int l15  = lane & 15;
  const int wm   = (wave & 3) * 32;     // wave m-offset in block tile
  const int wn   = (wave >> 2) * 64;    // wave n-offset in block tile

  v8f acc[2][4] = {};

  // staging: each thread moves 16 halves of A and 16 halves of B per K-step
  const int lrow = tid >> 1;            // 0..127
  const int lcol = (tid & 1) * 16;      // 0 or 16 (halves)
  const _Float16* xp = X + (size_t)(bm + lrow) * ldx + lcol;
  const _Float16* yp = Y + (size_t)(bn + lrow) * ldy + lcol;

  // prologue: fetch K-step 0 into registers
  v8h ra0 = ((const v8h*)xp)[0], ra1 = ((const v8h*)xp)[1];
  v8h rb0 = ((const v8h*)yp)[0], rb1 = ((const v8h*)yp)[1];

  for (int k0 = 0; k0 < K; k0 += 32) {
    __syncthreads();
    v8h* ad = (v8h*)(As + lrow * 32 + lcol);
    v8h* bd = (v8h*)(Bs + lrow * 32 + lcol);
    ad[0] = ra0; ad[1] = ra1;
    bd[0] = rb0; bd[1] = rb1;
    __syncthreads();

    // prefetch next K-step (overlaps with the 8 WMMAs below)
    if (k0 + 32 < K) {
      const v8h* xn = (const v8h*)(xp + k0 + 32);
      const v8h* yn = (const v8h*)(yp + k0 + 32);
      ra0 = xn[0]; ra1 = xn[1];
      rb0 = yn[0]; rb1 = yn[1];
    }

    // batch all fragment loads so LDS latency overlaps the WMMA chain
    HFrag af[2], bf[4];
#pragma unroll
    for (int r = 0; r < 2; ++r) {
      const v8h* arow = (const v8h*)(As + (wm + r * 16 + l15) * 32);
      af[r].h[0] = arow[hi];       // K chunks {hi, 2+hi}
      af[r].h[1] = arow[2 + hi];
    }
#pragma unroll
    for (int s = 0; s < 4; ++s) {
      const v8h* brow = (const v8h*)(Bs + (wn + s * 16 + l15) * 32);
      bf[s].h[0] = brow[hi];       // K chunks {hi, 2+hi}
      bf[s].h[1] = brow[2 + hi];
    }
#pragma unroll
    for (int r = 0; r < 2; ++r)
#pragma unroll
      for (int s = 0; s < 4; ++s)
        acc[r][s] = __builtin_amdgcn_wmma_f32_16x16x32_f16(
            false, af[r].v, false, bf[s].v, (short)0, acc[r][s], false, false);
  }

#pragma unroll
  for (int r = 0; r < 2; ++r)
#pragma unroll
    for (int s = 0; s < 4; ++s)
#pragma unroll
      for (int q = 0; q < 8; ++q) {
        int row = bm + wm + r * 16 + q + hi * 8;
        int col = bn + wn + s * 16 + l15;
        C[(size_t)row * ldc + col] = acc[r][s][q];
      }
}

// ---------------------------------------------------------------------------
// Column mean / inv-std (population std, matches jnp.std ddof=0)
// ---------------------------------------------------------------------------
__global__ __launch_bounds__(256)
void colstats_kernel(const float* __restrict__ X, float* mean, float* inv,
                     int M, int N) {
  int c = blockIdx.x * blockDim.x + threadIdx.x;
  if (c >= N) return;
  float s = 0.f, ss = 0.f;
  for (int r = 0; r < M; ++r) {
    float v = X[(size_t)r * N + c];
    s += v; ss += v * v;
  }
  float m   = s / (float)M;
  float var = fmaxf(ss / (float)M - m * m, 0.f);
  mean[c] = m;
  inv[c]  = 1.f / (sqrtf(var) + 1e-8f);
}

__global__ __launch_bounds__(256)
void znorm_cast_kernel(const float* __restrict__ X, const float* __restrict__ mean,
                       const float* __restrict__ inv, _Float16* __restrict__ Z,
                       size_t total, int N) {
  for (size_t i = blockIdx.x * (size_t)blockDim.x + threadIdx.x; i < total;
       i += (size_t)gridDim.x * blockDim.x) {
    int c = (int)(i % N);
    Z[i] = (_Float16)((X[i] - mean[c]) * inv[c]);
  }
}

__global__ __launch_bounds__(256)
void cast_f16_kernel(const float* __restrict__ X, _Float16* __restrict__ Z, size_t total) {
  for (size_t i = blockIdx.x * (size_t)blockDim.x + threadIdx.x; i < total;
       i += (size_t)gridDim.x * blockDim.x)
    Z[i] = (_Float16)X[i];
}

// W: K x N (f32) -> WT: N x K (f16)
__global__ __launch_bounds__(256)
void transpose_cast_kernel(const float* __restrict__ W, _Float16* __restrict__ WT,
                           int K, int N) {
  size_t total = (size_t)K * N;
  for (size_t i = blockIdx.x * (size_t)blockDim.x + threadIdx.x; i < total;
       i += (size_t)gridDim.x * blockDim.x) {
    int n = (int)(i / K), k = (int)(i % K);
    WT[i] = (_Float16)W[(size_t)k * N + n];
  }
}

__global__ __launch_bounds__(256)
void row_sumsq_f16_kernel(const _Float16* __restrict__ Z, float* __restrict__ out, int N) {
  __shared__ float buf[256];
  int r = blockIdx.x;
  float s = 0.f;
  for (int c = threadIdx.x; c < N; c += 256) {
    float v = (float)Z[(size_t)r * N + c];
    s += v * v;
  }
  buf[threadIdx.x] = s; __syncthreads();
  for (int st = 128; st > 0; st >>= 1) {
    if (threadIdx.x < st) buf[threadIdx.x] += buf[threadIdx.x + st];
    __syncthreads();
  }
  if (threadIdx.x == 0) out[r] = buf[0];
}

__global__ __launch_bounds__(256)
void rowsum_f32_kernel(const float* __restrict__ X, float* __restrict__ out, int N) {
  __shared__ float buf[256];
  int r = blockIdx.x;
  float s = 0.f;
  for (int c = threadIdx.x; c < N; c += 256) s += X[(size_t)r * N + c];
  buf[threadIdx.x] = s; __syncthreads();
  for (int st = 128; st > 0; st >>= 1) {
    if (threadIdx.x < st) buf[threadIdx.x] += buf[threadIdx.x + st];
    __syncthreads();
  }
  if (threadIdx.x == 0) out[r] = buf[0];
}

// ---------------------------------------------------------------------------
// Elementwise kernel transforms
// ---------------------------------------------------------------------------
__global__ __launch_bounds__(256)
void gene_transform_kernel(float* __restrict__ Kg, const float* __restrict__ sq, int n) {
  size_t total = (size_t)n * n;
  for (size_t e = blockIdx.x * (size_t)blockDim.x + threadIdx.x; e < total;
       e += (size_t)gridDim.x * blockDim.x) {
    int i = (int)(e / n), j = (int)(e % n);
    float d2 = fmaxf(sq[i] + sq[j] - 2.f * Kg[e], 0.f);
    Kg[e] = expf(-d2 * (1.f / 8192.f));  // 2*SIGMA^2 = 8192
  }
}

__global__ __launch_bounds__(256)
void cna_transform_kernel(float* __restrict__ Kc, size_t total) {
  for (size_t e = blockIdx.x * (size_t)blockDim.x + threadIdx.x; e < total;
       e += (size_t)gridDim.x * blockDim.x) {
    float v = Kc[e] + 1.f;
    Kc[e] = v * v * v;
  }
}

__global__ __launch_bounds__(256)
void jaccard_kernel(float* __restrict__ K, const float* __restrict__ s, int n) {
  size_t total = (size_t)n * n;
  for (size_t e = blockIdx.x * (size_t)blockDim.x + threadIdx.x; e < total;
       e += (size_t)gridDim.x * blockDim.x) {
    int i = (int)(e / n), j = (int)(e % n);
    float inter = K[e];
    K[e] = inter / (s[i] + s[j] - inter + 1e-8f);
  }
}

__global__ __launch_bounds__(256)
void frob_kernel(const float* __restrict__ X, size_t total, float* __restrict__ out) {
  __shared__ float buf[256];
  float s = 0.f;
  for (size_t e = blockIdx.x * (size_t)blockDim.x + threadIdx.x; e < total;
       e += (size_t)gridDim.x * blockDim.x) {
    float v = X[e]; s += v * v;
  }
  buf[threadIdx.x] = s; __syncthreads();
  for (int st = 128; st > 0; st >>= 1) {
    if (threadIdx.x < st) buf[threadIdx.x] += buf[threadIdx.x + st];
    __syncthreads();
  }
  if (threadIdx.x == 0) atomicAdd(out, buf[0]);
}

// shared = (ge+bg + ce+bc + me+bm)/3
__global__ __launch_bounds__(256)
void shared_kernel(const float* ge, const float* ce, const float* me,
                   const float* bg, const float* bc, const float* bm,
                   float* __restrict__ out, size_t total) {
  for (size_t e = blockIdx.x * (size_t)blockDim.x + threadIdx.x; e < total;
       e += (size_t)gridDim.x * blockDim.x) {
    int c = (int)(e & (SDIM - 1));
    out[e] = (ge[e] + bg[c] + ce[e] + bc[c] + me[e] + bm[c]) * (1.f / 3.f);
  }
}

__global__ __launch_bounds__(256)
void addbias_kernel(float* __restrict__ X, const float* __restrict__ b, size_t total) {
  for (size_t e = blockIdx.x * (size_t)blockDim.x + threadIdx.x; e < total;
       e += (size_t)gridDim.x * blockDim.x)
    X[e] += b[(int)(e & (SDIM - 1))];
}

// ---------------------------------------------------------------------------
// Tiny MLPs: per-row 128->32->3 softmax (accumulate mean) and 128->32->1 sigmoid
// ---------------------------------------------------------------------------
__global__ __launch_bounds__(64)
void att_dyn_kernel(const float* __restrict__ Xs, const float* __restrict__ W1,
                    const float* __restrict__ b1, const float* __restrict__ W2,
                    const float* __restrict__ b2, float* __restrict__ dyn_sum) {
  __shared__ float xr[SDIM];
  __shared__ float hid[ADIM];
  __shared__ float lg[3];
  int i = blockIdx.x, t = threadIdx.x;
  xr[t] = Xs[(size_t)i * SDIM + t];
  xr[t + 64] = Xs[(size_t)i * SDIM + t + 64];
  __syncthreads();
  if (t < ADIM) {
    float a = b1[t];
    for (int s = 0; s < SDIM; ++s) a += xr[s] * W1[s * ADIM + t];
    hid[t] = fmaxf(a, 0.f);
  }
  __syncthreads();
  if (t < 3) {
    float a = b2[t];
    for (int h = 0; h < ADIM; ++h) a += hid[h] * W2[h * 3 + t];
    lg[t] = a;
  }
  __syncthreads();
  if (t == 0) {
    float mx = fmaxf(lg[0], fmaxf(lg[1], lg[2]));
    float e0 = expf(lg[0] - mx), e1 = expf(lg[1] - mx), e2 = expf(lg[2] - mx);
    float s = e0 + e1 + e2;
    atomicAdd(&dyn_sum[0], e0 / s);
    atomicAdd(&dyn_sum[1], e1 / s);
    atomicAdd(&dyn_sum[2], e2 / s);
  }
}

__global__ __launch_bounds__(64)
void sig_att_kernel(const float* __restrict__ Xs, const float* __restrict__ W1,
                    const float* __restrict__ b1, const float* __restrict__ W2,
                    const float* __restrict__ b2, float* __restrict__ out) {
  __shared__ float xr[SDIM];
  __shared__ float hid[ADIM];
  int i = blockIdx.x, t = threadIdx.x;
  xr[t] = Xs[(size_t)i * SDIM + t];
  xr[t + 64] = Xs[(size_t)i * SDIM + t + 64];
  __syncthreads();
  if (t < ADIM) {
    float a = b1[t];
    for (int s = 0; s < SDIM; ++s) a += xr[s] * W1[s * ADIM + t];
    hid[t] = fmaxf(a, 0.f);
  }
  __syncthreads();
  if (t == 0) {
    float a = b2[0];
    for (int h = 0; h < ADIM; ++h) a += hid[h] * W2[h];
    out[i] = 1.f / (1.f + expf(-a));
  }
}

// sc[0..2]=fro sums, sc[3..5]=dyn sums -> sc[8..10] = mixed_k / (||K_k||_F + 1e-8)
__global__ void finalize_scalars_kernel(const float* __restrict__ static_w,
                                        float* __restrict__ sc, float inv_rows) {
  if (blockIdx.x != 0 || threadIdx.x != 0) return;
  float mx = fmaxf(static_w[0], fmaxf(static_w[1], static_w[2]));
  float e0 = expf(static_w[0] - mx), e1 = expf(static_w[1] - mx), e2 = expf(static_w[2] - mx);
  float s = e0 + e1 + e2;
  float st[3] = {e0 / s, e1 / s, e2 / s};
  for (int k = 0; k < 3; ++k) {
    float mixed = 0.5f * st[k] + 0.5f * sc[3 + k] * inv_rows;
    sc[8 + k] = mixed / (sqrtf(sc[k]) + 1e-8f);
  }
}

__global__ __launch_bounds__(256)
void fuse_kernel(float* __restrict__ Kg, const float* __restrict__ Kc,
                 const float* __restrict__ Km, const float* __restrict__ sc,
                 size_t total) {
  float c0 = sc[8], c1 = sc[9], c2 = sc[10];
  for (size_t e = blockIdx.x * (size_t)blockDim.x + threadIdx.x; e < total;
       e += (size_t)gridDim.x * blockDim.x)
    Kg[e] = c0 * Kg[e] + c1 * Kc[e] + c2 * Km[e];
}

// ---------------------------------------------------------------------------
// Per-row top-k (k=10) of S[i,:] * att[:]; scatter original S values.
// Tie-break: lowest index (matches jax top_k).
// ---------------------------------------------------------------------------
__global__ __launch_bounds__(256)
void topk_kernel(const float* __restrict__ S, const float* __restrict__ att, int n,
                 float* __restrict__ Sout, int* __restrict__ idx_out,
                 float* __restrict__ val_out) {
  __shared__ float w[2048];
  __shared__ float rv[256];
  __shared__ int   ri[256];
  __shared__ int   sel[KTOP];
  int i = blockIdx.x;
  const float* row = S + (size_t)i * n;
  for (int j = threadIdx.x; j < n; j += 256) w[j] = row[j] * att[j];
  __syncthreads();
  for (int t = 0; t < KTOP; ++t) {
    float bv = -INFINITY; int bi = n;
    for (int j = threadIdx.x; j < n; j += 256) {
      float v = w[j];
      if (v > bv || (v == bv && j < bi)) { bv = v; bi = j; }
    }
    rv[threadIdx.x] = bv; ri[threadIdx.x] = bi;
    __syncthreads();
    for (int st = 128; st > 0; st >>= 1) {
      if (threadIdx.x < st) {
        float ov = rv[threadIdx.x + st]; int oi = ri[threadIdx.x + st];
        if (ov > rv[threadIdx.x] || (ov == rv[threadIdx.x] && oi < ri[threadIdx.x])) {
          rv[threadIdx.x] = ov; ri[threadIdx.x] = oi;
        }
      }
      __syncthreads();
    }
    if (threadIdx.x == 0) { sel[t] = ri[0]; w[ri[0]] = -INFINITY; }
    __syncthreads();
  }
  for (int j = threadIdx.x; j < n; j += 256) Sout[(size_t)i * n + j] = 0.f;
  __syncthreads();
  if (threadIdx.x < KTOP) {
    int j = sel[threadIdx.x];
    float v = row[j];
    Sout[(size_t)i * n + j] = v;
    idx_out[i * KTOP + threadIdx.x] = j;
    val_out[i * KTOP + threadIdx.x] = v;
  }
}

__global__ __launch_bounds__(256)
void drug_dinv_kernel(const float* __restrict__ vals, float* __restrict__ dinv, int R) {
  int i = blockIdx.x * blockDim.x + threadIdx.x;
  if (i >= R) return;
  float s = 0.f;
  for (int t = 0; t < KTOP; ++t) s += vals[i * KTOP + t];
  dinv[i] = 1.f / sqrtf(s + 1e-9f);
}

__global__ __launch_bounds__(256)
void scale_dense_kernel(float* __restrict__ S, const float* __restrict__ dinv, int n) {
  size_t total = (size_t)n * n;
  for (size_t e = blockIdx.x * (size_t)blockDim.x + threadIdx.x; e < total;
       e += (size_t)gridDim.x * blockDim.x) {
    int i = (int)(e / n), j = (int)(e % n);
    S[e] *= dinv[i] * dinv[j];
  }
}

__global__ __launch_bounds__(256)
void scale_vals_kernel(float* __restrict__ vals, const int* __restrict__ idx,
                       const float* __restrict__ dinv, int R) {
  int e = blockIdx.x * blockDim.x + threadIdx.x;
  if (e >= R * KTOP) return;
  vals[e] *= dinv[e / KTOP] * dinv[idx[e]];
}

// feat[i,:] = bias + sum_t vals[i,t] * W[idx[i,t], :]
__global__ __launch_bounds__(SDIM)
void gather_feat_kernel(const int* __restrict__ idx, const float* __restrict__ vals,
                        const float* __restrict__ W, const float* __restrict__ bias,
                        float* __restrict__ out) {
  int i = blockIdx.x, d = threadIdx.x;
  float acc = bias[d];
  for (int t = 0; t < KTOP; ++t)
    acc += vals[i * KTOP + t] * W[(size_t)idx[i * KTOP + t] * SDIM + d];
  out[(size_t)i * SDIM + d] = acc;
}

__global__ __launch_bounds__(256)
void cd_sums_kernel(const float* __restrict__ cd, float* __restrict__ rs,
                    float* __restrict__ cs) {
  size_t total = (size_t)NCELL * NDRUG;
  for (size_t e = blockIdx.x * (size_t)blockDim.x + threadIdx.x; e < total;
       e += (size_t)gridDim.x * blockDim.x) {
    if (cd[e] > 0.9f) {
      atomicAdd(&rs[e / NDRUG], 1.f);
      atomicAdd(&cs[e % NDRUG], 1.f);
    }
  }
}

__global__ __launch_bounds__(256)
void dinvA_kernel(const float* __restrict__ valc, const float* __restrict__ vald,
                  const float* __restrict__ rs_cd, const float* __restrict__ cs_cd,
                  float* __restrict__ dinvA) {
  int i = blockIdx.x * blockDim.x + threadIdx.x;
  if (i >= NTOT) return;
  float s;
  if (i < NCELL) {
    s = rs_cd[i];
    for (int t = 0; t < KTOP; ++t) s += valc[i * KTOP + t];
  } else {
    int d = i - NCELL;
    s = cs_cd[d];
    for (int t = 0; t < KTOP; ++t) s += vald[d * KTOP + t];
  }
  dinvA[i] = 1.f / sqrtf(s + 1e-9f);
}

__global__ __launch_bounds__(256)
void adjhat_kernel(const float* __restrict__ filt_cell, const float* __restrict__ filt_drug,
                   const float* __restrict__ cd, const float* __restrict__ dinvA,
                   float* __restrict__ out) {
  size_t total = (size_t)NTOT * NTOT;
  for (size_t e = blockIdx.x * (size_t)blockDim.x + threadIdx.x; e < total;
       e += (size_t)gridDim.x * blockDim.x) {
    int i = (int)(e / NTOT), j = (int)(e % NTOT);
    float a;
    if (i < NCELL) {
      if (j < NCELL) a = filt_cell[(size_t)i * NCELL + j];
      else           a = (cd[(size_t)i * NDRUG + (j - NCELL)] > 0.9f) ? 1.f : 0.f;
    } else {
      if (j < NCELL) a = (cd[(size_t)j * NDRUG + (i - NCELL)] > 0.9f) ? 1.f : 0.f;
      else           a = filt_drug[(size_t)(i - NCELL) * NDRUG + (j - NCELL)];
    }
    float v = dinvA[i] * a * dinvA[j];
    out[e] = (i == j) ? v + 1.f : v;
  }
}

__global__ __launch_bounds__(SDIM)
void fnorm_kernel(const float* __restrict__ feats, float* __restrict__ fn) {
  __shared__ float buf[SDIM];
  int i = blockIdx.x;
  float v = feats[(size_t)i * SDIM + threadIdx.x];
  buf[threadIdx.x] = v * v;
  __syncthreads();
  for (int st = 64; st > 0; st >>= 1) {
    if (threadIdx.x < st) buf[threadIdx.x] += buf[threadIdx.x + st];
    __syncthreads();
  }
  fn[(size_t)i * SDIM + threadIdx.x] = v / (sqrtf(buf[0]) + 1e-8f);
}

// InfoNCE per row: pos/neg from sign of off-diagonal adj_hat (dinv > 0 keeps sign of A)
__global__ __launch_bounds__(256)
void loss_kernel(const float* __restrict__ fn, const float* __restrict__ adj,
                 float* __restrict__ sc) {
  __shared__ float fi[SDIM];
  __shared__ float rp[256], rn[256], rc[256];
  int i = blockIdx.x;
  if (threadIdx.x < SDIM) fi[threadIdx.x] = fn[(size_t)i * SDIM + threadIdx.x];
  __syncthreads();
  float sp = 0.f, sn = 0.f, cp = 0.f;
  for (int j = threadIdx.x; j < NTOT; j += 256) {
    if (j == i) continue;
    const float* fj = fn + (size_t)j * SDIM;
    float d = 0.f;
    for (int t = 0; t < SDIM; ++t) d += fi[t] * fj[t];
    float e = expf(d * 2.0f);  // 1/TAU = 2
    if (adj[(size_t)i * NTOT + j] > 0.f) { sp += e; cp += 1.f; }
    else                                 { sn += e; }
  }
  rp[threadIdx.x] = sp; rn[threadIdx.x] = sn; rc[threadIdx.x] = cp;
  __syncthreads();
  for (int st = 128; st > 0; st >>= 1) {
    if (threadIdx.x < st) {
      rp[threadIdx.x] += rp[threadIdx.x + st];
      rn[threadIdx.x] += rn[threadIdx.x + st];
      rc[threadIdx.x] += rc[threadIdx.x + st];
    }
    __syncthreads();
  }
  if (threadIdx.x == 0) {
    float denom = rp[0] + rn[0] + 1e-9f;
    if (rc[0] > 1e-9f && denom > 1e-8f) {
      float ratio = fmaxf(rp[0] / denom, 1e-9f);
      atomicAdd(&sc[6], -logf(ratio));
      atomicAdd(&sc[7], 1.f);
    }
  }
}

__global__ void loss_final_kernel(const float* __restrict__ sc, float* __restrict__ out) {
  if (blockIdx.x == 0 && threadIdx.x == 0) out[0] = sc[6] / fmaxf(sc[7], 1.f);
}

// ---------------------------------------------------------------------------
extern "C" void kernel_launch(void* const* d_in, const int* in_sizes, int n_in,
                              void* d_out, int out_size, void* d_ws, size_t ws_size,
                              hipStream_t stream) {
  const float* gene      = (const float*)d_in[0];
  const float* cna       = (const float*)d_in[1];
  const float* mutation  = (const float*)d_in[2];
  const float* fdrug     = (const float*)d_in[3];
  const float* cd_adj    = (const float*)d_in[4];
  const float* Wg        = (const float*)d_in[5];
  const float* bg        = (const float*)d_in[6];
  const float* Wc        = (const float*)d_in[7];
  const float* bc        = (const float*)d_in[8];
  const float* Wm        = (const float*)d_in[9];
  const float* bm        = (const float*)d_in[10];
  const float* Wd        = (const float*)d_in[11];
  const float* bd        = (const float*)d_in[12];
  const float* attW1     = (const float*)d_in[13];
  const float* attb1     = (const float*)d_in[14];
  const float* attW2     = (const float*)d_in[15];
  const float* attb2     = (const float*)d_in[16];
  const float* caW1      = (const float*)d_in[17];
  const float* cab1      = (const float*)d_in[18];
  const float* caW2      = (const float*)d_in[19];
  const float* cab2      = (const float*)d_in[20];
  const float* daW1      = (const float*)d_in[21];
  const float* dab1      = (const float*)d_in[22];
  const float* daW2      = (const float*)d_in[23];
  const float* dab2      = (const float*)d_in[24];
  const float* kcW       = (const float*)d_in[25];
  const float* kcb       = (const float*)d_in[26];
  const float* kdW       = (const float*)d_in[27];
  const float* kdb       = (const float*)d_in[28];
  const float* static_w  = (const float*)d_in[29];

  // ---- workspace bump allocator --------------------------------------------
  char* ws = (char*)d_ws;
  size_t off = 0;
  auto alloc = [&](size_t bytes) -> void* {
    void* p = ws + off;
    off = (off + bytes + 255) & ~(size_t)255;
    return p;
  };

  _Float16* zg16   = (_Float16*)alloc((size_t)NCELL * DGENE * 2);
  _Float16* zc16   = (_Float16*)alloc((size_t)NCELL * DCNA * 2);
  _Float16* mut16  = (_Float16*)alloc((size_t)NCELL * DMUT * 2);
  _Float16* drug16 = (_Float16*)alloc((size_t)NDRUG * DDRUG * 2);
  _Float16* WgT    = (_Float16*)alloc((size_t)SDIM * DGENE * 2);
  _Float16* WcT    = (_Float16*)alloc((size_t)SDIM * DCNA * 2);
  _Float16* WmT    = (_Float16*)alloc((size_t)SDIM * DMUT * 2);
  _Float16* WdT    = (_Float16*)alloc((size_t)SDIM * DDRUG * 2);
  float* Kg        = (float*)alloc((size_t)NCELL * NCELL * 4);   // -> fused kernel
  float* Kc        = (float*)alloc((size_t)NCELL * NCELL * 4);   // -> filt_cell (aliased)
  float* Km        = (float*)alloc((size_t)NCELL * NCELL * 4);   // -> filt_drug (aliased)
  float* Kd        = (float*)alloc((size_t)NDRUG * NDRUG * 4);
  float* ge        = (float*)alloc((size_t)NCELL * SDIM * 4);
  float* ce        = (float*)alloc((size_t)NCELL * SDIM * 4);
  float* me        = (float*)alloc((size_t)NCELL * SDIM * 4);
  float* de        = (float*)alloc((size_t)NDRUG * SDIM * 4);
  float* sharedb   = (float*)alloc((size_t)NCELL * SDIM * 4);
  float* feats     = (float*)alloc((size_t)NTOT * SDIM * 4);
  float* fnb       = (float*)alloc((size_t)NTOT * SDIM * 4);
  float* meang     = (float*)alloc(DGENE * 4);
  float* invg      = (float*)alloc(DGENE * 4);
  float* meanc     = (float*)alloc(DCNA * 4);
  float* invc      = (float*)alloc(DCNA * 4);
  float* sqg       = (float*)alloc(NCELL * 4);
  float* smut      = (float*)alloc(NCELL * 4);
  float* sdrug     = (float*)alloc(NDRUG * 4);
  float* catt      = (float*)alloc(NCELL * 4);
  float* datt      = (float*)alloc(NDRUG * 4);
  int*   idxc      = (int*)alloc((size_t)NCELL * KTOP * 4);
  float* valc      = (float*)alloc((size_t)NCELL * KTOP * 4);
  int*   idxd      = (int*)alloc((size_t)NDRUG * KTOP * 4);
  float* vald      = (float*)alloc((size_t)NDRUG * KTOP * 4);
  float* dinvd     = (float*)alloc(NDRUG * 4);
  float* dinvA     = (float*)alloc(NTOT * 4);
  // zero-initialized zone: 64 scalars + rs_cd + cs_cd
  size_t zeroFloats = 64 + NCELL + NDRUG;
  float* zone  = (float*)alloc(zeroFloats * 4);
  float* sc    = zone;                 // [0..2] fro, [3..5] dyn, [6..7] loss, [8..10] coef
  float* rs_cd = zone + 64;
  float* cs_cd = zone + 64 + NCELL;

  float* filt_cell = Kc;   // Kc dead after fuse
  float* filt_drug = Km;   // Km dead after fuse

  float* out_adj  = (float*)d_out;
  float* out_loss = out_adj + (size_t)NTOT * NTOT;

  // ---- 0. zero accumulators -------------------------------------------------
  hipMemsetAsync(zone, 0, zeroFloats * 4, stream);

  // ---- 1. normalization / casts --------------------------------------------
  colstats_kernel<<<dim3((DGENE + 255) / 256), 256, 0, stream>>>(gene, meang, invg, NCELL, DGENE);
  colstats_kernel<<<dim3((DCNA + 255) / 256), 256, 0, stream>>>(cna, meanc, invc, NCELL, DCNA);
  znorm_cast_kernel<<<2048, 256, 0, stream>>>(gene, meang, invg, zg16, (size_t)NCELL * DGENE, DGENE);
  znorm_cast_kernel<<<2048, 256, 0, stream>>>(cna, meanc, invc, zc16, (size_t)NCELL * DCNA, DCNA);
  cast_f16_kernel<<<2048, 256, 0, stream>>>(mutation, mut16, (size_t)NCELL * DMUT);
  cast_f16_kernel<<<1024, 256, 0, stream>>>(fdrug, drug16, (size_t)NDRUG * DDRUG);
  transpose_cast_kernel<<<1024, 256, 0, stream>>>(Wg, WgT, DGENE, SDIM);
  transpose_cast_kernel<<<512, 256, 0, stream>>>(Wc, WcT, DCNA, SDIM);
  transpose_cast_kernel<<<512, 256, 0, stream>>>(Wm, WmT, DMUT, SDIM);
  transpose_cast_kernel<<<512, 256, 0, stream>>>(Wd, WdT, DDRUG, SDIM);

  row_sumsq_f16_kernel<<<NCELL, 256, 0, stream>>>(zg16, sqg, DGENE);
  rowsum_f32_kernel<<<NCELL, 256, 0, stream>>>(mutation, smut, DMUT);
  rowsum_f32_kernel<<<NDRUG, 256, 0, stream>>>(fdrug, sdrug, DDRUG);

  // ---- 2. WMMA gram matrices and embeddings --------------------------------
  gemm_xyt_f16<<<dim3(NCELL / 128, NCELL / 128), 256, 0, stream>>>(
      zg16, zg16, Kg, NCELL, NCELL, DGENE, DGENE, DGENE, NCELL);
  gemm_xyt_f16<<<dim3(NCELL / 128, NCELL / 128), 256, 0, stream>>>(
      zc16, zc16, Kc, NCELL, NCELL, DCNA, DCNA, DCNA, NCELL);
  gemm_xyt_f16<<<dim3(NCELL / 128, NCELL / 128), 256, 0, stream>>>(
      mut16, mut16, Km, NCELL, NCELL, DMUT, DMUT, DMUT, NCELL);
  gemm_xyt_f16<<<dim3(NDRUG / 128, NDRUG / 128), 256, 0, stream>>>(
      drug16, drug16, Kd, NDRUG, NDRUG, DDRUG, DDRUG, DDRUG, NDRUG);
  gemm_xyt_f16<<<dim3(SDIM / 128, NCELL / 128), 256, 0, stream>>>(
      zg16, WgT, ge, NCELL, SDIM, DGENE, DGENE, DGENE, SDIM);
  gemm_xyt_f16<<<dim3(SDIM / 128, NCELL / 128), 256, 0, stream>>>(
      zc16, WcT, ce, NCELL, SDIM, DCNA, DCNA, DCNA, SDIM);
  gemm_xyt_f16<<<dim3(SDIM / 128, NCELL / 128), 256, 0, stream>>>(
      mut16, WmT, me, NCELL, SDIM, DMUT, DMUT, DMUT, SDIM);
  gemm_xyt_f16<<<dim3(SDIM / 128, NDRUG / 128), 256, 0, stream>>>(
      drug16, WdT, de, NDRUG, SDIM, DDRUG, DDRUG, DDRUG, SDIM);

  // ---- 3. kernel transforms + Frobenius norms ------------------------------
  gene_transform_kernel<<<4096, 256, 0, stream>>>(Kg, sqg, NCELL);
  cna_transform_kernel<<<4096, 256, 0, stream>>>(Kc, (size_t)NCELL * NCELL);
  jaccard_kernel<<<4096, 256, 0, stream>>>(Km, smut, NCELL);
  jaccard_kernel<<<1024, 256, 0, stream>>>(Kd, sdrug, NDRUG);
  frob_kernel<<<2048, 256, 0, stream>>>(Kg, (size_t)NCELL * NCELL, &sc[0]);
  frob_kernel<<<2048, 256, 0, stream>>>(Kc, (size_t)NCELL * NCELL, &sc[1]);
  frob_kernel<<<2048, 256, 0, stream>>>(Km, (size_t)NCELL * NCELL, &sc[2]);

  // ---- 4. fusion + attention MLPs ------------------------------------------
  shared_kernel<<<1024, 256, 0, stream>>>(ge, ce, me, bg, bc, bm, sharedb, (size_t)NCELL * SDIM);
  addbias_kernel<<<512, 256, 0, stream>>>(de, bd, (size_t)NDRUG * SDIM);
  att_dyn_kernel<<<NCELL, 64, 0, stream>>>(sharedb, attW1, attb1, attW2, attb2, &sc[3]);
  sig_att_kernel<<<NCELL, 64, 0, stream>>>(sharedb, caW1, cab1, caW2, cab2, catt);
  sig_att_kernel<<<NDRUG, 64, 0, stream>>>(de, daW1, dab1, daW2, dab2, datt);
  finalize_scalars_kernel<<<1, 1, 0, stream>>>(static_w, sc, 1.f / (float)NCELL);
  fuse_kernel<<<4096, 256, 0, stream>>>(Kg, Kc, Km, sc, (size_t)NCELL * NCELL);

  // ---- 5. top-k neighbor filters (att normalization is rank-invariant) -----
  topk_kernel<<<NCELL, 256, 0, stream>>>(Kg, catt, NCELL, filt_cell, idxc, valc);
  topk_kernel<<<NDRUG, 256, 0, stream>>>(Kd, datt, NDRUG, filt_drug, idxd, vald);

  // ---- 6. drug symmetric normalization -------------------------------------
  drug_dinv_kernel<<<(NDRUG + 255) / 256, 256, 0, stream>>>(vald, dinvd, NDRUG);
  scale_dense_kernel<<<1024, 256, 0, stream>>>(filt_drug, dinvd, NDRUG);
  scale_vals_kernel<<<(NDRUG * KTOP + 255) / 256, 256, 0, stream>>>(vald, idxd, dinvd, NDRUG);

  // ---- 7. node features via sparse gather-GEMM -----------------------------
  gather_feat_kernel<<<NCELL, SDIM, 0, stream>>>(idxc, valc, kcW, kcb, feats);
  gather_feat_kernel<<<NDRUG, SDIM, 0, stream>>>(idxd, vald, kdW, kdb, feats + (size_t)NCELL * SDIM);

  // ---- 8. block adjacency, degree normalization, adj_hat -> d_out ----------
  cd_sums_kernel<<<2048, 256, 0, stream>>>(cd_adj, rs_cd, cs_cd);
  dinvA_kernel<<<(NTOT + 255) / 256, 256, 0, stream>>>(valc, vald, rs_cd, cs_cd, dinvA);
  adjhat_kernel<<<4096, 256, 0, stream>>>(filt_cell, filt_drug, cd_adj, dinvA, out_adj);

  // ---- 9. InfoNCE loss ------------------------------------------------------
  fnorm_kernel<<<NTOT, SDIM, 0, stream>>>(feats, fnb);
  loss_kernel<<<NTOT, 256, 0, stream>>>(fnb, out_adj, sc);
  loss_final_kernel<<<1, 1, 0, stream>>>(sc, out_loss);
}
